// APPNP_40415642255917
// MI455X (gfx1250) — compile-verified, matrix operand order checked by reading, not run on hardware
//
#include <hip/hip_runtime.h>
#include <hip/hip_fp16.h>

typedef __attribute__((ext_vector_type(16))) _Float16 v16h;
typedef __attribute__((ext_vector_type(8)))  float    v8f;

static constexpr int IN_F  = 256;
static constexpr int HID_F = 128;
static constexpr int OUT_F = 64;

// ---------------- utility kernels ----------------

__global__ void k_fill0(float4* __restrict__ p, long n4) {
  long t = (long)blockIdx.x * blockDim.x + threadIdx.x;
  if (t < n4) p[t] = make_float4(0.f, 0.f, 0.f, 0.f);
}

__global__ void k_degrees(const int* __restrict__ src, const int* __restrict__ dst,
                          float* __restrict__ degO, float* __restrict__ degI, int E) {
  int t = blockIdx.x * blockDim.x + threadIdx.x;
  if (t >= E) return;
  unsafeAtomicAdd(&degO[src[t]], 1.0f);
  unsafeAtomicAdd(&degI[dst[t]], 1.0f);
}

__global__ void k_rsqrt_clip(float* __restrict__ p, int n) {
  int t = blockIdx.x * blockDim.x + threadIdx.x;
  if (t < n) p[t] = rsqrtf(fmaxf(p[t], 1.0f));
}

// convert f32 weight [K x C] to f16 transposed [C x K] so GEMM B-fragment
// loads become contiguous 16-byte chunks (global_load_b128).
__global__ void k_cvt_f16_T(const float* __restrict__ in, _Float16* __restrict__ out,
                            int K, int C) {
  int t = blockIdx.x * blockDim.x + threadIdx.x;
  if (t >= K * C) return;
  int k = t / C;
  int c = t - k * C;
  out[(size_t)c * K + k] = (_Float16)in[t];
}

// ---------------- WMMA GEMM ----------------
// C[M x ncols] = A[M x K](f32 -> f16 on the fly) * B (f16, stored transposed ncols x K),
// fp32 accumulate via V_WMMA_F32_16X16X32_F16. One wave computes NT adjacent
// 16x16 column tiles of one 16-row strip: A fragment loaded once per K-step,
// reused for NT WMMAs.
// A fragment (ISA 16-bit A layout): lanes 0-15 -> row M=lane, K {0..7,16..23};
// lanes 16-31 -> same rows, K {8..15,24..31}. B symmetric by column.
// C layout: element r, lanes 0-15 -> M=r; lanes 16-31 -> M=r+8; N = lane&15.

template <int NT>
__global__ __launch_bounds__(256) void k_gemm_wmma(
    const float* __restrict__ A, const _Float16* __restrict__ Bt, float* __restrict__ C,
    int colGroups, int totGroups, int K, int ncols) {
  int wave = (int)((blockIdx.x * blockDim.x + threadIdx.x) >> 5);
  if (wave >= totGroups) return;
  int rt = wave / colGroups;
  int cg = wave - rt * colGroups;
  int lane = threadIdx.x & 31;
  int half = lane >> 4;
  int l15  = lane & 15;
  const float* arow = A + (size_t)(rt * 16 + l15) * K;

  v8f acc[NT];
#pragma unroll
  for (int j = 0; j < NT; ++j) acc[j] = (v8f){};

  for (int k0 = 0; k0 < K; k0 += 32) {
    int ka = k0 + half * 8;
    v16h a;
#pragma unroll
    for (int i = 0; i < 8; ++i) {
      a[i]     = (_Float16)arow[ka + i];
      a[i + 8] = (_Float16)arow[ka + 16 + i];
    }
#pragma unroll
    for (int j = 0; j < NT; ++j) {
      const _Float16* bp = Bt + (size_t)((cg * NT + j) * 16 + l15) * K;
      v16h b;
#pragma unroll
      for (int i = 0; i < 8; ++i) {      // two contiguous 16B chunks
        b[i]     = bp[ka + i];
        b[i + 8] = bp[ka + 16 + i];
      }
      acc[j] = __builtin_amdgcn_wmma_f32_16x16x32_f16(
          /*neg_a=*/false, a, /*neg_b=*/false, b,
          /*c_mod=*/(short)0, acc[j], /*reuse_a=*/false, /*reuse_b=*/false);
    }
  }
#pragma unroll
  for (int j = 0; j < NT; ++j) {
    float* crow = C + (size_t)(rt * 16 + half * 8) * ncols + (cg * NT + j) * 16 + l15;
#pragma unroll
    for (int r = 0; r < 8; ++r)
      crow[(size_t)r * ncols] = acc[j][r];
  }
}

// ---------------- edge scatter: agg[dst] += x[src] * ns[src] ----------------
// One wave per edge; lane covers D/32 contiguous floats (float4 gather at D=128).

template <int LOGD>
__global__ __launch_bounds__(256) void k_edge_spmm(
    const float* __restrict__ x, const float* __restrict__ ns,
    const int* __restrict__ src, const int* __restrict__ dst,
    float* __restrict__ agg, int E) {
  int wid = (int)(((long)blockIdx.x * blockDim.x + threadIdx.x) >> 5);
  if (wid >= E) return;
  int lane = threadIdx.x & 31;
  constexpr int PER = (1 << LOGD) >> 5;
  int s = src[wid];
  int d = dst[wid];
  float w = ns[s];
  const float* xs = x + ((size_t)s << LOGD) + lane * PER;
  float* ad = agg + ((size_t)d << LOGD) + lane * PER;
  if constexpr (PER == 4) {
    float4 v = *reinterpret_cast<const float4*>(xs);
    unsafeAtomicAdd(&ad[0], v.x * w);
    unsafeAtomicAdd(&ad[1], v.y * w);
    unsafeAtomicAdd(&ad[2], v.z * w);
    unsafeAtomicAdd(&ad[3], v.w * w);
  } else {
    float2 v = *reinterpret_cast<const float2*>(xs);
    unsafeAtomicAdd(&ad[0], v.x * w);
    unsafeAtomicAdd(&ad[1], v.y * w);
  }
}

// out[t] = in[t] * nd[row] (+ bias[col])
__global__ void k_scale_rows(const float* __restrict__ in, const float* __restrict__ nd,
                             const float* __restrict__ bias, float* __restrict__ out,
                             long total, int logD) {
  long t = (long)blockIdx.x * blockDim.x + threadIdx.x;
  if (t >= total) return;
  long row = t >> logD;
  int col = (int)(t & ((1 << logD) - 1));
  float v = in[t] * nd[row];
  if (bias) v += bias[col];
  out[t] = v;
}

// ---------------- host orchestration ----------------

extern "C" void kernel_launch(void* const* d_in, const int* in_sizes, int n_in,
                              void* d_out, int out_size, void* d_ws, size_t ws_size,
                              hipStream_t stream) {
  const float* feat = (const float*)d_in[0];
  const int*   src  = (const int*)d_in[1];
  const int*   dst  = (const int*)d_in[2];
  const float* W1   = (const float*)d_in[3];
  const float* b1   = (const float*)d_in[4];
  const float* W3   = (const float*)d_in[5];
  const float* b3   = (const float*)d_in[6];
  (void)n_in; (void)out_size; (void)ws_size;

  const int N = in_sizes[0] / IN_F;   // 50000 (multiple of 16)
  const int E = in_sizes[1];          // 800000

  char* ws = (char*)d_ws;
  size_t cur = 0;
  auto wsalloc = [&](size_t bytes) -> void* {
    void* p = ws + cur;
    cur = (cur + bytes + 255) & ~(size_t)255;
    return p;
  };
  float*    norms = (float*)wsalloc((size_t)2 * N * sizeof(float)); // [ns | nd]
  float*    ns    = norms;
  float*    nd    = norms + N;
  _Float16* w1t   = (_Float16*)wsalloc((size_t)IN_F * HID_F * sizeof(_Float16)); // [HID x IN]
  _Float16* w3t   = (_Float16*)wsalloc((size_t)HID_F * OUT_F * sizeof(_Float16)); // [OUT x HID]
  float*    bufA  = (float*)wsalloc((size_t)N * HID_F * sizeof(float));
  float*    bufB  = (float*)wsalloc((size_t)N * HID_F * sizeof(float));
  float*    bufC  = (float*)wsalloc((size_t)N * OUT_F * sizeof(float));

  const int TB = 256;
  auto blocks = [](long n, int tb) { return (int)((n + tb - 1) / tb); };

  // 1) degrees -> symmetric norms
  k_fill0<<<blocks((long)2 * N / 4, TB), TB, 0, stream>>>((float4*)norms, (long)2 * N / 4);
  k_degrees<<<blocks(E, TB), TB, 0, stream>>>(src, dst, ns, nd, E);
  k_rsqrt_clip<<<blocks((long)2 * N, TB), TB, 0, stream>>>(norms, 2 * N);

  // 2) weights -> f16, transposed (once)
  k_cvt_f16_T<<<blocks(IN_F * HID_F, TB), TB, 0, stream>>>(W1, w1t, IN_F, HID_F);
  k_cvt_f16_T<<<blocks(HID_F * OUT_F, TB), TB, 0, stream>>>(W3, w3t, HID_F, OUT_F);

  // 3) X1 = feat @ W1 -> bufA   (WMMA, 4 col tiles per wave)
  {
    int colGroups = (HID_F / 16) / 4;              // 2
    int totGroups = (N / 16) * colGroups;          // 6250 waves
    k_gemm_wmma<4><<<blocks((long)totGroups * 32, TB), TB, 0, stream>>>(
        feat, w1t, bufA, colGroups, totGroups, IN_F, HID_F);
  }

  auto prop128 = [&](const float* in, float* out) {
    k_fill0<<<blocks((long)N * HID_F / 4, TB), TB, 0, stream>>>((float4*)out, (long)N * HID_F / 4);
    k_edge_spmm<7><<<blocks((long)E * 32, TB), TB, 0, stream>>>(in, ns, src, dst, out, E);
  };

  // 4) GraphConv1: h = prop(X1)*nd + b1 -> bufB
  prop128(bufA, bufB);
  k_scale_rows<<<blocks((long)N * HID_F, TB), TB, 0, stream>>>(bufB, nd, b1, bufB, (long)N * HID_F, 7);

  // 5) APPNP: 4 power iterations, alpha = 0  -> h ends in bufB
  float* cur_h = bufB;
  float* alt   = bufA;
  for (int it = 0; it < 4; ++it) {
    prop128(cur_h, alt);
    k_scale_rows<<<blocks((long)N * HID_F, TB), TB, 0, stream>>>(alt, nd, nullptr, alt, (long)N * HID_F, 7);
    float* tmp = cur_h; cur_h = alt; alt = tmp;
  }

  // 6) X3 = h @ W3 -> bufC   (WMMA, 4 col tiles per wave covers all of OUT)
  {
    int colGroups = (OUT_F / 16) / 4;              // 1
    int totGroups = (N / 16) * colGroups;          // 3125 waves
    k_gemm_wmma<4><<<blocks((long)totGroups * 32, TB), TB, 0, stream>>>(
        cur_h, w3t, bufC, colGroups, totGroups, HID_F, OUT_F);
  }

  // 7) final prop at D=64, epilogue writes d_out: out = prop(X3)*nd + b3
  k_fill0<<<blocks((long)N * OUT_F / 4, TB), TB, 0, stream>>>((float4*)alt, (long)N * OUT_F / 4);
  k_edge_spmm<6><<<blocks((long)E * 32, TB), TB, 0, stream>>>(bufC, ns, src, dst, alt, E);
  k_scale_rows<<<blocks((long)N * OUT_F, TB), TB, 0, stream>>>(alt, nd, b3, (float*)d_out, (long)N * OUT_F, 6);
}
